// ASGCNN_55233279427068
// MI455X (gfx1250) — matrix operand order repaired and checked by reading, not compiled, not run on hardware
//
#include <hip/hip_runtime.h>
#include <hip/hip_bf16.h>
#include <cstdint>
#include <cstddef>

// ---------------------------------------------------------------------------
// ASGCNN dual-branch CGCNN forward for MI455X (gfx1250, wave32, WMMA).
// Heavy per-edge dual MLPs run on v_wmma_f32_16x16x32_bf16 with fp32 accum.
// A-tiles are gathered into LDS with global_load_async_to_lds_b128 (ASYNCcnt).
// ---------------------------------------------------------------------------

#define INV_STD 0.9999950000374997f   // 1/sqrt(1+1e-5), BN eval with mean=0 var=1

typedef __attribute__((ext_vector_type(16))) __bf16 v16bf;
typedef __attribute__((ext_vector_type(8)))  __bf16 v8bf;
typedef __attribute__((ext_vector_type(8)))  float  v8f;

__device__ __forceinline__ float silu_f(float x) {
  return x / (1.0f + __expf(-x));
}
__device__ __forceinline__ float softplus_f(float x) {
  return (x > 20.0f) ? x : log1pf(__expf(x));
}

// ---------------------------------------------------------------------------
// fp32 -> bf16 conversion (weights, once per launch; deterministic)
// ---------------------------------------------------------------------------
__global__ void f32_to_bf16_kernel(const float* __restrict__ in,
                                   __bf16* __restrict__ out, size_t n) {
  size_t i = (size_t)blockIdx.x * blockDim.x + threadIdx.x;
  if (i < n) out[i] = (__bf16)in[i];
}

__global__ void fill_zero_kernel(float* __restrict__ p, size_t n) {
  size_t i = (size_t)blockIdx.x * blockDim.x + threadIdx.x;
  if (i < n) p[i] = 0.0f;
}

// ---------------------------------------------------------------------------
// Embedding MLP: h = silu(BN(v @ W)), K = 92.  One block per node, D threads.
// ---------------------------------------------------------------------------
template <int D>
__global__ __launch_bounds__(D)
void embed_kernel(const float* __restrict__ v, const float* __restrict__ W,
                  const float* __restrict__ b, const float* __restrict__ g,
                  const float* __restrict__ bt,
                  float* __restrict__ h, __bf16* __restrict__ hbf) {
  __shared__ float vr[92];
  const int n = blockIdx.x, j = threadIdx.x;
  for (int k = j; k < 92; k += D) vr[k] = v[(size_t)n * 92 + k];
  __syncthreads();
  float acc = 0.0f;
  #pragma unroll 4
  for (int k = 0; k < 92; ++k) acc += vr[k] * W[k * D + j];
  const float y = silu_f((acc + b[j]) * (g[j] * INV_STD) + bt[j]);
  h[(size_t)n * D + j]   = y;
  hbf[(size_t)n * D + j] = (__bf16)y;
}

// ---------------------------------------------------------------------------
// Edge MLP: hm = silu(BN(e @ eW)), K = 41.  One block per edge, D threads.
// ---------------------------------------------------------------------------
template <int D>
__global__ __launch_bounds__(D)
void edge_mlp_kernel(const float* __restrict__ ef, const float* __restrict__ W,
                     const float* __restrict__ b, const float* __restrict__ g,
                     const float* __restrict__ bt,
                     __bf16* __restrict__ hm) {
  __shared__ float er[41];
  const int e = blockIdx.x, j = threadIdx.x;
  for (int k = j; k < 41; k += D) er[k] = ef[(size_t)e * 41 + k];
  __syncthreads();
  float acc = 0.0f;
  #pragma unroll
  for (int k = 0; k < 41; ++k) acc += er[k] * W[k * D + j];
  const float y = silu_f((acc + b[j]) * (g[j] * INV_STD) + bt[j]);
  hm[(size_t)e * D + j] = (__bf16)y;
}

// ---------------------------------------------------------------------------
// Heavy conv kernel: 32-edge M-tile, A = [h[src] | h[dst] | hm]  (32 x 3D bf16)
//   u = silu(BN(A @ mW)), s = softplus(BN(A @ sW)), scatter-add u*s to agg[dst].
// 128 threads = 4 wave32; each wave owns D/64 N-tiles; per K-step each B
// fragment (gate + value) is reused for both 16-row A fragments -> 4 WMMAs per
// 2 B loads (halves per-edge L2 weight traffic vs a 16-edge tile).
// A staged into LDS with per-lane global_load_async_to_lds_b128 (ASYNCcnt).
// Fragment layouts per ISA 7.12.2:
//   A 16x32 bf16 : lane<16 -> K {kb..kb+7, kb+16..kb+23} of row lane&15;
//                  lane>=16 -> the complementary K octets.
//   B 32x16 bf16 : col = lane&15, 16 K in A's octet order (column of row-major weights).
//   D 16x16 f32  : col = lane&15, rows r + 8*(lane>>4).
// ---------------------------------------------------------------------------
template <int D>
__global__ __launch_bounds__(128)
void conv_edge_wmma(const __bf16* __restrict__ hbf,  // [N, D]
                    const __bf16* __restrict__ hm,   // [E, D] (this layer)
                    const int* __restrict__ src, const int* __restrict__ dst,
                    const __bf16* __restrict__ mW,   // [3D, D] bf16, this layer
                    const __bf16* __restrict__ sW,   // [3D, D] bf16, this layer
                    const float* __restrict__ pmb, const float* __restrict__ pmg,
                    const float* __restrict__ pmbt,
                    const float* __restrict__ psb, const float* __restrict__ psg,
                    const float* __restrict__ psbt,
                    float* __restrict__ agg, int E) {
  constexpr int K3 = 3 * D;
  constexpr int NT = D / 64;            // N-tiles per wave (D=128 -> 2, D=64 -> 1)
  __shared__ alignas(32) __bf16 Ash[32][K3];
  __shared__ int sIdx[32], dIdx[32];

  const int tid = threadIdx.x;
  const int e0  = blockIdx.x * 32;

  if (tid < 32) {
    const int e  = e0 + tid;
    const int ec = (e < E) ? e : (E - 1);
    sIdx[tid] = src[ec];
    dIdx[tid] = dst[ec];
  }
  __syncthreads();

  // ---- async gather of the 32 x 3D A-tile into LDS --------------------
  // 16-byte chunks; 32*CH is a multiple of 128 -> uniform trip count.
  constexpr int CH = K3 / 8;
  const uint32_t ldsA = (uint32_t)(uintptr_t)&Ash[0][0];
  for (int c = tid; c < 32 * CH; c += 128) {
    const int row = c / CH;
    const int cc  = c % CH;
    const int seg = cc / (D / 8);
    const int o8  = (cc % (D / 8)) * 8;
    const int ec  = ((e0 + row) < E) ? (e0 + row) : (E - 1);
    const __bf16* p;
    if (seg == 0)      p = hbf + (size_t)sIdx[row] * D + o8;
    else if (seg == 1) p = hbf + (size_t)dIdx[row] * D + o8;
    else               p = hm  + (size_t)ec * D + o8;
    const uint32_t dsto = ldsA + (uint32_t)(row * K3 + cc * 8) * 2u;
    const unsigned long long gaddr = (unsigned long long)p;
    asm volatile("global_load_async_to_lds_b128 %0, %1, off"
                 :: "v"(dsto), "v"(gaddr) : "memory");
  }
  asm volatile("s_wait_asynccnt 0x0" ::: "memory");
  __syncthreads();

  const int wave = tid >> 5;
  const int lane = tid & 31;
  const int mrow = lane & 15;
  const int kg   = (lane >> 4) << 3;    // 0 or 8

  union AccT { v8f v; float f[8]; };
  AccT accU[2][NT], accS[2][NT];
  #pragma unroll
  for (int hh = 0; hh < 2; ++hh)
    #pragma unroll
    for (int t = 0; t < NT; ++t)
      #pragma unroll
      for (int i = 0; i < 8; ++i) { accU[hh][t].f[i] = 0.0f; accS[hh][t].f[i] = 0.0f; }

  for (int kb = 0; kb < K3; kb += 32) {
    union { v16bf v; v8bf h[2]; } af0, af1;
    af0.h[0] = *(const v8bf*)&Ash[mrow][kb + kg];
    af0.h[1] = *(const v8bf*)&Ash[mrow][kb + 16 + kg];
    af1.h[0] = *(const v8bf*)&Ash[16 + mrow][kb + kg];
    af1.h[1] = *(const v8bf*)&Ash[16 + mrow][kb + 16 + kg];
    #pragma unroll
    for (int t = 0; t < NT; ++t) {
      const int ncol0 = (wave * NT + t) * 16;
      v16bf bm, bs;
      #pragma unroll
      for (int s = 0; s < 16; ++s) {
        const int kk = kb + ((s < 8) ? (kg + s) : (16 + kg + s - 8));
        const size_t wi = (size_t)kk * D + ncol0 + mrow;
        bm[s] = mW[wi];
        bs[s] = sW[wi];
      }
      accU[0][t].v = __builtin_amdgcn_wmma_f32_16x16x32_bf16(
          false, af0.v, false, bm, (short)0, accU[0][t].v, false, false);
      accU[1][t].v = __builtin_amdgcn_wmma_f32_16x16x32_bf16(
          false, af1.v, false, bm, (short)0, accU[1][t].v, false, false);
      accS[0][t].v = __builtin_amdgcn_wmma_f32_16x16x32_bf16(
          false, af0.v, false, bs, (short)0, accS[0][t].v, false, false);
      accS[1][t].v = __builtin_amdgcn_wmma_f32_16x16x32_bf16(
          false, af1.v, false, bs, (short)0, accS[1][t].v, false, false);
    }
  }

  // Epilogue: BN + silu/softplus gate, scatter-add to agg[dst].
  const int nlane = lane & 15;
  const int mbase = (lane >> 4) << 3;
  #pragma unroll
  for (int t = 0; t < NT; ++t) {
    const int ncol = (wave * NT + t) * 16 + nlane;
    const float gm = pmg[ncol] * INV_STD, bm_ = pmb[ncol], btm = pmbt[ncol];
    const float gs = psg[ncol] * INV_STD, bs_ = psb[ncol], bts = psbt[ncol];
    #pragma unroll
    for (int hh = 0; hh < 2; ++hh) {
      #pragma unroll
      for (int r = 0; r < 8; ++r) {
        const int m = hh * 16 + mbase + r;
        const int e = e0 + m;
        if (e < E) {
          const float u = silu_f((accU[hh][t].f[r] + bm_) * gm + btm);
          const float s = softplus_f((accS[hh][t].f[r] + bs_) * gs + bts);
          atomicAdd(&agg[(size_t)dIdx[m] * D + ncol], u * s);
        }
      }
    }
  }
}

// ---------------------------------------------------------------------------
// Node update: h = softplus(agg * g' + b + h); refresh fp32 + bf16 copies.
// ---------------------------------------------------------------------------
template <int D>
__global__ void node_update_kernel(const float* __restrict__ agg,
                                   const float* __restrict__ ng,
                                   const float* __restrict__ nb,
                                   float* __restrict__ h, __bf16* __restrict__ hbf,
                                   int N) {
  const size_t i = (size_t)blockIdx.x * blockDim.x + threadIdx.x;
  if (i >= (size_t)N * D) return;
  const int j = (int)(i % D);
  const float y = softplus_f(agg[i] * (ng[j] * INV_STD) + nb[j] + h[i]);
  h[i]   = y;
  hbf[i] = (__bf16)y;
}

// ---------------------------------------------------------------------------
// Segment-sum pooling (+ node counts) via atomics.
// ---------------------------------------------------------------------------
template <int D>
__global__ void pool_sum_kernel(const float* __restrict__ h, const int* __restrict__ gid,
                                float* __restrict__ pooled, float* __restrict__ cnt,
                                int N) {
  const size_t i = (size_t)blockIdx.x * blockDim.x + threadIdx.x;
  if (i >= (size_t)N * D) return;
  const int n = (int)(i / D), j = (int)(i % D);
  const int g = gid[n];
  atomicAdd(&pooled[(size_t)g * D + j], h[i]);
  if (j == 0) atomicAdd(&cnt[g], 1.0f);
}

// ---------------------------------------------------------------------------
// Head: concat[64+128] -> 128 -> 64 -> 1. One block per graph (tiny GEMV).
// ---------------------------------------------------------------------------
__global__ __launch_bounds__(128)
void head_kernel(const float* __restrict__ pooled_a, const float* __restrict__ cnt_a,
                 const float* __restrict__ pooled_s, const float* __restrict__ cnt_s,
                 const float* __restrict__ f1W, const float* __restrict__ f1b,
                 const float* __restrict__ f1g, const float* __restrict__ f1bt,
                 const float* __restrict__ f2W, const float* __restrict__ f2b,
                 const float* __restrict__ f2g, const float* __restrict__ f2bt,
                 const float* __restrict__ pW, const float* __restrict__ pb,
                 float* __restrict__ out) {
  __shared__ float vin[192];
  __shared__ float t1[128];
  __shared__ float t2[64];
  const int g = blockIdx.x, t = threadIdx.x;
  const float ca = fmaxf(cnt_a[g], 1.0f), cs = fmaxf(cnt_s[g], 1.0f);
  if (t < 64)  vin[t]      = pooled_a[g * 64 + t] / ca;
  vin[64 + t] = pooled_s[g * 128 + t] / cs;
  __syncthreads();
  {
    float acc = 0.0f;
    #pragma unroll 4
    for (int k = 0; k < 192; ++k) acc += vin[k] * f1W[k * 128 + t];
    t1[t] = silu_f((acc + f1b[t]) * (f1g[t] * INV_STD) + f1bt[t]);
  }
  __syncthreads();
  if (t < 64) {
    float acc = 0.0f;
    #pragma unroll 4
    for (int k = 0; k < 128; ++k) acc += t1[k] * f2W[k * 64 + t];
    t2[t] = silu_f((acc + f2b[t]) * (f2g[t] * INV_STD) + f2bt[t]);
  }
  __syncthreads();
  if (t == 0) {
    float acc = 0.0f;
    #pragma unroll
    for (int k = 0; k < 64; ++k) acc += t2[k] * pW[k];
    out[g] = acc + pb[0];
  }
}

// ---------------------------------------------------------------------------
// Host-side orchestration.
// Input order = setup_inputs() dict order; nested param dicts flatten per JAX
// pytree rules (keys sorted alphabetically).
// ---------------------------------------------------------------------------
extern "C" void kernel_launch(void* const* d_in, const int* in_sizes, int n_in,
                              void* d_out, int out_size, void* d_ws, size_t ws_size,
                              hipStream_t stream) {
  (void)in_sizes; (void)n_in; (void)out_size; (void)ws_size;
  constexpr int NA = 2000,  EA = 16000,  DA = 64;
  constexpr int NS = 30000, ES = 300000, DS = 128;
  constexpr int G = 256, FE = 41, L = 3;

  const float* va    = (const float*)d_in[0];
  const float* ea    = (const float*)d_in[1];
  const int*   src_a = (const int*)d_in[2];
  const int*   dst_a = (const int*)d_in[3];
  const int*   gid_a = (const int*)d_in[4];
  const float* vs    = (const float*)d_in[5];
  const float* es    = (const float*)d_in[6];
  const int*   src_s = (const int*)d_in[7];
  const int*   dst_s = (const int*)d_in[8];
  const int*   gid_s = (const int*)d_in[9];

  auto P = [&](int i) { return (const float*)d_in[i]; };
  // params_a (base 10): eW eb ebt eg embW embb embbt embg mW mb mbt mg nb ng sW sb sbt sg
  const float *a_eW = P(10), *a_eb = P(11), *a_ebt = P(12), *a_eg = P(13),
              *a_embW = P(14), *a_embb = P(15), *a_embbt = P(16), *a_embg = P(17),
              *a_mW = P(18), *a_mb = P(19), *a_mbt = P(20), *a_mg = P(21),
              *a_nb = P(22), *a_ng = P(23),
              *a_sW = P(24), *a_sb = P(25), *a_sbt = P(26), *a_sg = P(27);
  // params_s (base 28)
  const float *s_eW = P(28), *s_eb = P(29), *s_ebt = P(30), *s_eg = P(31),
              *s_embW = P(32), *s_embb = P(33), *s_embbt = P(34), *s_embg = P(35),
              *s_mW = P(36), *s_mb = P(37), *s_mbt = P(38), *s_mg = P(39),
              *s_nb = P(40), *s_ng = P(41),
              *s_sW = P(42), *s_sb = P(43), *s_sbt = P(44), *s_sg = P(45);
  // params_h (base 46): f1W f1b f1bt f1g f2W f2b f2bt f2g pW pb
  const float *f1W = P(46), *f1b = P(47), *f1bt = P(48), *f1g = P(49),
              *f2W = P(50), *f2b = P(51), *f2bt = P(52), *f2g = P(53),
              *pWt = P(54), *pbt = P(55);

  // ---- workspace carve (256-B aligned) --------------------------------
  uint8_t* wbase = (uint8_t*)d_ws;
  size_t off = 0;
  auto alloc = [&](size_t bytes) -> void* {
    off = (off + 255) & ~(size_t)255;
    void* p = wbase + off;
    off += bytes;
    return p;
  };
  float*  h_a     = (float*) alloc((size_t)NA * DA * 4);
  __bf16* hbf_a   = (__bf16*)alloc((size_t)NA * DA * 2);
  float*  agg_a   = (float*) alloc((size_t)NA * DA * 4);
  __bf16* hm_a    = (__bf16*)alloc((size_t)EA * DA * 2);
  __bf16* mWbf_a  = (__bf16*)alloc((size_t)L * 3 * DA * DA * 2);
  __bf16* sWbf_a  = (__bf16*)alloc((size_t)L * 3 * DA * DA * 2);
  float*  h_s     = (float*) alloc((size_t)NS * DS * 4);
  __bf16* hbf_s   = (__bf16*)alloc((size_t)NS * DS * 2);
  float*  agg_s   = (float*) alloc((size_t)NS * DS * 4);
  __bf16* hm_s    = (__bf16*)alloc((size_t)ES * DS * 2);
  __bf16* mWbf_s  = (__bf16*)alloc((size_t)L * 3 * DS * DS * 2);
  __bf16* sWbf_s  = (__bf16*)alloc((size_t)L * 3 * DS * DS * 2);
  float*  pool_a  = (float*) alloc((size_t)G * DA * 4);
  float*  pool_s  = (float*) alloc((size_t)G * DS * 4);
  float*  cnt_a   = (float*) alloc((size_t)G * 4);
  float*  cnt_s   = (float*) alloc((size_t)G * 4);

  // ---- weight conversion to bf16 --------------------------------------
  {
    const size_t na = (size_t)L * 3 * DA * DA;
    f32_to_bf16_kernel<<<(unsigned)((na + 255) / 256), 256, 0, stream>>>(a_mW, mWbf_a, na);
    f32_to_bf16_kernel<<<(unsigned)((na + 255) / 256), 256, 0, stream>>>(a_sW, sWbf_a, na);
    const size_t ns = (size_t)L * 3 * DS * DS;
    f32_to_bf16_kernel<<<(unsigned)((ns + 255) / 256), 256, 0, stream>>>(s_mW, mWbf_s, ns);
    f32_to_bf16_kernel<<<(unsigned)((ns + 255) / 256), 256, 0, stream>>>(s_sW, sWbf_s, ns);
  }

  // ---- embedding ------------------------------------------------------
  embed_kernel<DA><<<NA, DA, 0, stream>>>(va, a_embW, a_embb, a_embg, a_embbt, h_a, hbf_a);
  embed_kernel<DS><<<NS, DS, 0, stream>>>(vs, s_embW, s_embb, s_embg, s_embbt, h_s, hbf_s);

  // ---- conv layers ----------------------------------------------------
  for (int l = 0; l < L; ++l) {
    // adsorbate branch (D = 64)
    edge_mlp_kernel<DA><<<EA, DA, 0, stream>>>(ea, a_eW + (size_t)l * FE * DA,
                                               a_eb + l * DA, a_eg + l * DA,
                                               a_ebt + l * DA, hm_a);
    {
      const size_t n = (size_t)NA * DA;
      fill_zero_kernel<<<(unsigned)((n + 255) / 256), 256, 0, stream>>>(agg_a, n);
    }
    conv_edge_wmma<DA><<<(EA + 31) / 32, 128, 0, stream>>>(
        hbf_a, hm_a, src_a, dst_a,
        mWbf_a + (size_t)l * 3 * DA * DA, sWbf_a + (size_t)l * 3 * DA * DA,
        a_mb + l * DA, a_mg + l * DA, a_mbt + l * DA,
        a_sb + l * DA, a_sg + l * DA, a_sbt + l * DA, agg_a, EA);
    {
      const size_t n = (size_t)NA * DA;
      node_update_kernel<DA><<<(unsigned)((n + 255) / 256), 256, 0, stream>>>(
          agg_a, a_ng + l * DA, a_nb + l * DA, h_a, hbf_a, NA);
    }

    // slab branch (D = 128)
    edge_mlp_kernel<DS><<<ES, DS, 0, stream>>>(es, s_eW + (size_t)l * FE * DS,
                                               s_eb + l * DS, s_eg + l * DS,
                                               s_ebt + l * DS, hm_s);
    {
      const size_t n = (size_t)NS * DS;
      fill_zero_kernel<<<(unsigned)((n + 255) / 256), 256, 0, stream>>>(agg_s, n);
    }
    conv_edge_wmma<DS><<<(ES + 31) / 32, 128, 0, stream>>>(
        hbf_s, hm_s, src_s, dst_s,
        mWbf_s + (size_t)l * 3 * DS * DS, sWbf_s + (size_t)l * 3 * DS * DS,
        s_mb + l * DS, s_mg + l * DS, s_mbt + l * DS,
        s_sb + l * DS, s_sg + l * DS, s_sbt + l * DS, agg_s, ES);
    {
      const size_t n = (size_t)NS * DS;
      node_update_kernel<DS><<<(unsigned)((n + 255) / 256), 256, 0, stream>>>(
          agg_s, s_ng + l * DS, s_nb + l * DS, h_s, hbf_s, NS);
    }
  }

  // ---- pooling --------------------------------------------------------
  {
    size_t n = (size_t)G * DA;
    fill_zero_kernel<<<(unsigned)((n + 255) / 256), 256, 0, stream>>>(pool_a, n);
    n = (size_t)G * DS;
    fill_zero_kernel<<<(unsigned)((n + 255) / 256), 256, 0, stream>>>(pool_s, n);
    fill_zero_kernel<<<1, 256, 0, stream>>>(cnt_a, (size_t)G);
    fill_zero_kernel<<<1, 256, 0, stream>>>(cnt_s, (size_t)G);
  }
  {
    const size_t n = (size_t)NA * DA;
    pool_sum_kernel<DA><<<(unsigned)((n + 255) / 256), 256, 0, stream>>>(h_a, gid_a, pool_a, cnt_a, NA);
  }
  {
    const size_t n = (size_t)NS * DS;
    pool_sum_kernel<DS><<<(unsigned)((n + 255) / 256), 256, 0, stream>>>(h_s, gid_s, pool_s, cnt_s, NS);
  }

  // ---- head -----------------------------------------------------------
  head_kernel<<<G, 128, 0, stream>>>(pool_a, cnt_a, pool_s, cnt_s,
                                     f1W, f1b, f1g, f1bt,
                                     f2W, f2b, f2g, f2bt,
                                     pWt, pbt, (float*)d_out);
}
